// GraphAttentionDecoder_15960098472520
// MI455X (gfx1250) — matrix operand order repaired and checked by reading, not compiled, not run on hardware
//
#include <hip/hip_runtime.h>
#include <hip/hip_bf16.h>
#include <cstdint>
#include <cstddef>

// Problem constants (match reference): B=128, N=1000, D=512, H=8, V=8, DH=64
// P layout: [B*N, 1536] f32, columns = [K(0:512) | V(512:1024) | K_tanh(1024:1536)]

typedef __attribute__((ext_vector_type(16))) __bf16 v16bf;
typedef __attribute__((ext_vector_type(8)))  float  v8f;
typedef __attribute__((ext_vector_type(4)))  int    v4i;   // matches builtin param type

#if defined(__AMDGCN__) && __has_builtin(__builtin_amdgcn_global_load_async_to_lds_b128) && __has_builtin(__builtin_amdgcn_s_wait_asynccnt)
#define USE_ASYNC_LDS 1
#else
#define USE_ASYNC_LDS 0
#endif

static __device__ __forceinline__ __bf16 f2bf(float f) {
  union { float f; uint32_t u; } in; in.f = f;
  uint32_t r = in.u + 0x7FFFu + ((in.u >> 16) & 1u);   // round-to-nearest-even
  union { uint16_t s; __bf16 b; } out; out.s = (uint16_t)(r >> 16);
  return out.b;
}

// ---------------- pack kernels: f32 -> bf16 ----------------
__global__ void pack_emb(const float* __restrict__ src, __bf16* __restrict__ dst) {
  size_t i = ((size_t)blockIdx.x * blockDim.x + threadIdx.x) * 8;
  float4 a = ((const float4*)(src + i))[0];
  float4 b = ((const float4*)(src + i))[1];
  union { __bf16 h[8]; uint4 q; } o;
  o.h[0] = f2bf(a.x); o.h[1] = f2bf(a.y); o.h[2] = f2bf(a.z); o.h[3] = f2bf(a.w);
  o.h[4] = f2bf(b.x); o.h[5] = f2bf(b.y); o.h[6] = f2bf(b.z); o.h[7] = f2bf(b.w);
  *(uint4*)(dst + i) = o.q;
}

__global__ void pack_w(const float* __restrict__ Wk, const float* __restrict__ Wv,
                       const float* __restrict__ Wkt, __bf16* __restrict__ Wcat) {
  int gid = blockIdx.x * 256 + threadIdx.x;       // 3*512*512 total
  int w   = gid >> 18;                            // which weight
  int rem = gid & (262144 - 1);
  int k = rem >> 9, n = rem & 511;
  const float* src = (w == 0) ? Wk : (w == 1) ? Wv : Wkt;
  Wcat[(size_t)k * 1536 + w * 512 + n] = f2bf(src[(size_t)k * 512 + n]);
}

// ---------------- fused projection GEMM (bf16 WMMA, f32 accumulate) ----------------
// C[128000 x 1536] = A[128000 x 512] * Wcat[512 x 1536]
// block: 256 thr (8 waves), tile 128(M) x 128(N); wave tile 32x64 = 2x4 WMMA tiles.
__global__ __launch_bounds__(256) void gemm_bf16(
    const __bf16* __restrict__ A, const __bf16* __restrict__ Bw,
    float* __restrict__ P) {
  const int LD = 40;                       // padded K-stride in LDS (bf16 elems)
  __shared__ __bf16 lA[128 * 40];          // A tile, row-major [row][k]  (10.0 KB)
  __shared__ __bf16 lB[128 * 40];          // B tile, TRANSPOSED [n][k]   (10.0 KB)

  const int tid  = threadIdx.x;
  const int lane = tid & 31, wid = tid >> 5;
  const int wm = wid & 3, wn = wid >> 2;   // wave grid 4(M) x 2(N)
  const int lr = lane & 15;
  const int hi = (lane >> 4) & 1;          // upper half-wave
  const int m0 = blockIdx.y * 128;
  const int n0 = blockIdx.x * 128;

  v8f acc[2][4] = {};

  const int arow = tid >> 1, acol = (tid & 1) * 16;   // A loader: 16 bf16/thread
  const int bk = tid >> 3,   bn = (tid & 7) * 16;     // B loader: 16 bf16/thread

  for (int k0 = 0; k0 < 512; k0 += 32) {
    // --- stage A tile: byte-identical copy -> async global->LDS when available
    const __bf16* gA = A + (size_t)(m0 + arow) * 512 + k0 + acol;
#if USE_ASYNC_LDS
    __builtin_amdgcn_global_load_async_to_lds_b128(
        (v4i*)gA,       (v4i*)&lA[arow * LD + acol],     0, 0);
    __builtin_amdgcn_global_load_async_to_lds_b128(
        (v4i*)(gA + 8), (v4i*)&lA[arow * LD + acol + 8], 0, 0);
#else
    *(uint4*)&lA[arow * LD + acol]     = ((const uint4*)gA)[0];
    *(uint4*)&lA[arow * LD + acol + 8] = ((const uint4*)gA)[1];
#endif

    // --- stage B tile transposed (needs element shuffle -> manual path)
    union { uint4 q[2]; __bf16 h[16]; } bv;
    const uint4* gB = (const uint4*)(Bw + (size_t)(k0 + bk) * 1536 + n0 + bn);
    bv.q[0] = gB[0]; bv.q[1] = gB[1];
    #pragma unroll
    for (int i = 0; i < 16; ++i) lB[(bn + i) * LD + bk] = bv.h[i];

    if (k0 + 32 < 512)   // hint next A panel toward the WGP (global_prefetch_b8)
      __builtin_prefetch(gA + 32, 0, 3);

#if USE_ASYNC_LDS
    __builtin_amdgcn_s_wait_asynccnt(0);
#endif
    __syncthreads();

    // --- build fragments (ISA 7.12.2 16-bit layouts) and run 2x4 WMMA
    union F { v16bf v; uint4 q[2]; };
    F af[2], bf[4];
    const int kbA = hi ? 8 : 0;    // A: upper half-wave shifted by 8 in K
    const int kbB = hi ? 8 : 0;    // B: upper half-wave shifted by 8 in K
    #pragma unroll
    for (int i = 0; i < 2; ++i) {
      int base = (wm * 32 + i * 16 + lr) * LD + kbA;
      af[i].q[0] = *(const uint4*)&lA[base];        // K pairs {0,2,4,6}+kbA
      af[i].q[1] = *(const uint4*)&lA[base + 16];   // K pairs {16..22}+kbA
    }
    #pragma unroll
    for (int j = 0; j < 4; ++j) {
      int base = (wn * 64 + j * 16 + lr) * LD + kbB;
      bf[j].q[0] = *(const uint4*)&lB[base];        // K kbB+0..7
      bf[j].q[1] = *(const uint4*)&lB[base + 16];   // K kbB+16..23
    }
    #pragma unroll
    for (int i = 0; i < 2; ++i)
      #pragma unroll
      for (int j = 0; j < 4; ++j)
        acc[i][j] = __builtin_amdgcn_wmma_f32_16x16x32_bf16(
            false, af[i].v, false, bf[j].v, (short)0, acc[i][j], false, false);
    __syncthreads();
  }

  // D layout: VGPR r -> M=r (lanes 0-15) / M=8+r (lanes 16-31), N=lane%16
  #pragma unroll
  for (int i = 0; i < 2; ++i)
    #pragma unroll
    for (int j = 0; j < 4; ++j) {
      int row = m0 + wm * 32 + i * 16 + hi * 8;
      int col = n0 + wn * 64 + j * 16 + lr;
      #pragma unroll
      for (int r = 0; r < 8; ++r)
        P[(size_t)(row + r) * 1536 + col] = acc[i][j][r];
    }
}

// ---------------- Q = ctx @ Wq_ctx + step_ctx @ Wq_step ----------------
__global__ void q_kernel(const float* __restrict__ emb, const float* __restrict__ ctxv,
                         const int* __restrict__ prev_a, const float* __restrict__ dems,
                         const float* __restrict__ times, const float* __restrict__ Wq_ctx,
                         const float* __restrict__ Wq_step, float* __restrict__ Q) {
  int b = blockIdx.x, d = threadIdx.x;   // 512 threads
  float acc = 0.f;
  for (int k = 0; k < 512; ++k)
    acc += ctxv[b * 512 + k] * Wq_ctx[(size_t)k * 512 + d];
  for (int v = 0; v < 8; ++v) {
    int base = v * 514;
    int pa = prev_a[b * 8 + v];
    const float* e = emb + ((size_t)b * 1000 + pa) * 512;
    for (int i = 0; i < 512; ++i)
      acc += e[i] * Wq_step[(size_t)(base + i) * 512 + d];
    acc += dems[b * 8 + v]  * Wq_step[(size_t)(base + 512) * 512 + d];
    acc += times[b * 8 + v] * Wq_step[(size_t)(base + 513) * 512 + d];
  }
  Q[b * 512 + d] = acc;
}

// ---------------- masked MHA, seq_q = 1, one block per (b,h) ----------------
__global__ __launch_bounds__(256) void attn_kernel(const float* __restrict__ P,
    const float* __restrict__ Q, const unsigned char* __restrict__ mask,
    float* __restrict__ Ctx) {
  __shared__ float qs[64];
  __shared__ float lc[1000];
  __shared__ float red[256];
  int bh = blockIdx.x, b = bh >> 3, h = bh & 7;
  int tid = threadIdx.x;
  if (tid < 64) qs[tid] = Q[b * 512 + h * 64 + tid];
  __syncthreads();
  for (int n = tid; n < 1000; n += 256) {
    float c;
    if (mask[b * 1000 + n]) c = -1e9f;
    else {
      const float4* kp = (const float4*)(P + ((size_t)(b * 1000 + n)) * 1536 + h * 64);
      float dot = 0.f;
      #pragma unroll
      for (int i = 0; i < 16; ++i) {
        float4 kv = kp[i]; const float* qq = &qs[i * 4];
        dot += kv.x * qq[0] + kv.y * qq[1] + kv.z * qq[2] + kv.w * qq[3];
      }
      c = dot * 0.125f;                                   // 1/sqrt(64)
    }
    lc[n] = c;
  }
  __syncthreads();
  float lm = -3.4e38f;
  for (int n = tid; n < 1000; n += 256) lm = fmaxf(lm, lc[n]);
  red[tid] = lm; __syncthreads();
  for (int s = 128; s > 0; s >>= 1) { if (tid < s) red[tid] = fmaxf(red[tid], red[tid + s]); __syncthreads(); }
  float m = red[0]; __syncthreads();
  float ls = 0.f;
  for (int n = tid; n < 1000; n += 256) { float e = expf(lc[n] - m); lc[n] = e; ls += e; }
  red[tid] = ls; __syncthreads();
  for (int s = 128; s > 0; s >>= 1) { if (tid < s) red[tid] += red[tid + s]; __syncthreads(); }
  float inv = 1.f / red[0];
  int lane = tid & 31, w = tid >> 5;
  const float* vbase = P + (size_t)b * 1000 * 1536 + 512 + h * 64;
  for (int q = 0; q < 8; ++q) {           // each wave owns 8 head-dims
    int dh = w * 8 + q;
    float acc = 0.f;
    for (int n = lane; n < 1000; n += 32)
      acc += lc[n] * vbase[(size_t)n * 1536 + dh];
    #pragma unroll
    for (int off = 16; off > 0; off >>= 1) acc += __shfl_xor(acc, off);
    if (lane == 0) Ctx[b * 512 + h * 64 + dh] = acc * inv;
  }
}

// ---------------- mha = ctx @ W_out ----------------
__global__ void mha_kernel(const float* __restrict__ Ctx, const float* __restrict__ Wout,
                           float* __restrict__ Mha) {
  __shared__ float sc[512];
  int b = blockIdx.x, tid = threadIdx.x;
  for (int k = tid; k < 512; k += 256) sc[k] = Ctx[b * 512 + k];
  __syncthreads();
  for (int d = tid; d < 512; d += 256) {
    float acc = 0.f;
    for (int k = 0; k < 512; ++k) acc += sc[k] * Wout[(size_t)k * 512 + d];
    Mha[b * 512 + d] = acc;
  }
}

// ---------------- tanh-clipped logits + log_softmax ----------------
__global__ __launch_bounds__(256) void logits_kernel(const float* __restrict__ P,
    const float* __restrict__ Mha, const unsigned char* __restrict__ mask,
    float* __restrict__ out) {
  __shared__ float sm[512];
  __shared__ float lg[1000];
  __shared__ float red[256];
  int b = blockIdx.x, tid = threadIdx.x;
  for (int d = tid; d < 512; d += 256) sm[d] = Mha[b * 512 + d];
  __syncthreads();
  for (int n = tid; n < 1000; n += 256) {
    const float4* kp = (const float4*)(P + ((size_t)(b * 1000 + n)) * 1536 + 1024);
    float dot = 0.f;
    for (int i = 0; i < 128; ++i) {
      float4 kv = kp[i]; const float* qq = &sm[i * 4];
      dot += kv.x * qq[0] + kv.y * qq[1] + kv.z * qq[2] + kv.w * qq[3];
    }
    float l = tanhf(dot * 0.04419417382f) * 10.f;   // 1/sqrt(512), clip 10
    if (mask[b * 1000 + n]) l = -1e9f;
    lg[n] = l;
  }
  __syncthreads();
  float lm = -3.4e38f;
  for (int n = tid; n < 1000; n += 256) lm = fmaxf(lm, lg[n]);
  red[tid] = lm; __syncthreads();
  for (int s = 128; s > 0; s >>= 1) { if (tid < s) red[tid] = fmaxf(red[tid], red[tid + s]); __syncthreads(); }
  float m = red[0]; __syncthreads();
  float ls = 0.f;
  for (int n = tid; n < 1000; n += 256) ls += expf(lg[n] - m);
  red[tid] = ls; __syncthreads();
  for (int s = 128; s > 0; s >>= 1) { if (tid < s) red[tid] += red[tid + s]; __syncthreads(); }
  float lse = m + logf(red[0]);
  for (int n = tid; n < 1000; n += 256) out[(size_t)b * 1000 + n] = lg[n] - lse;
}

extern "C" void kernel_launch(void* const* d_in, const int* in_sizes, int n_in,
                              void* d_out, int out_size, void* d_ws, size_t ws_size,
                              hipStream_t stream) {
  const float* emb     = (const float*)d_in[0];   // [128,1000,512]
  const float* ctxv    = (const float*)d_in[1];   // [128,512]
  const int*   prev    = (const int*)d_in[2];     // [128,8]
  const float* dems    = (const float*)d_in[3];   // [128,8]
  const float* times   = (const float*)d_in[4];   // [128,8]
  const unsigned char* mask = (const unsigned char*)d_in[5]; // numpy bool: 1B/elem
  const float* Wq_ctx  = (const float*)d_in[6];   // [512,512]
  const float* Wq_step = (const float*)d_in[7];   // [4112,512]
  const float* Wk      = (const float*)d_in[8];
  const float* Wk_tanh = (const float*)d_in[9];
  const float* Wv      = (const float*)d_in[10];
  const float* Wout    = (const float*)d_in[11];
  float* out = (float*)d_out;                     // [128,1,1000] f32

  // workspace layout (all offsets 256B-aligned)
  char* ws = (char*)d_ws;
  __bf16* embB = (__bf16*)(ws);                                  // 131,072,000 B
  __bf16* Wcat = (__bf16*)(ws + 131072000ull);                   //   1,572,864 B
  float*  P    = (float*)(ws + 131072000ull + 1572864ull);       // 786,432,000 B
  float*  Q    = (float*)(ws + 131072000ull + 1572864ull + 786432000ull);
  float*  Ctx  = Q + 65536;
  float*  Mha  = Ctx + 65536;

  pack_emb <<<32000, 256, 0, stream>>>(emb, embB);
  pack_w   <<<3072,  256, 0, stream>>>(Wk, Wv, Wk_tanh, Wcat);
  gemm_bf16<<<dim3(12, 1000), 256, 0, stream>>>(embB, Wcat, P);
  q_kernel <<<128, 512, 0, stream>>>(emb, ctxv, prev, dems, times, Wq_ctx, Wq_step, Q);
  attn_kernel  <<<1024, 256, 0, stream>>>(P, Q, mask, Ctx);
  mha_kernel   <<<128, 256, 0, stream>>>(Ctx, Wout, Mha);
  logits_kernel<<<128, 256, 0, stream>>>(P, Mha, mask, out);
}
